// BasicModel_60722247631157
// MI455X (gfx1250) — matrix operand order repaired and checked by reading, not compiled, not run on hardware
//
#include <hip/hip_runtime.h>
#include <hip/hip_bf16.h>

// ---------------------------------------------------------------------------
// MI455X (gfx1250) fused SNN forward.
//   Phase 1: cur1 = x @ W1^T + b1 via v_wmma_f32_16x16x32_bf16, hi/lo bf16
//            split of both operands (3-term product) for near-fp32 accuracy.
//   Phase 2: 25-step Leaky-Integrate-and-Fire recurrence per batch row,
//            W2 staged transposed in LDS (uniform broadcast reads).
// Bandwidth-bound: ~103MB x-read + ~66MB output stores -> ~7us HBM floor.
// ---------------------------------------------------------------------------

typedef __attribute__((ext_vector_type(16))) __bf16 v16bf;
typedef __attribute__((ext_vector_type(8)))  float  v8f;

#define TSTEPS 25
#define NH 25          // hidden neurons
#define NO 10          // output neurons
#define KDIM 784
#define BETA 0.95f
#define ROWS_PER_BLOCK 256
#define LDS_STRIDE 26  // 25 cols padded -> bank-conflict-free (gcd(26,64)=2, 13 odd)

#define KK_ITERS 25    // ceil(784/32) K-chunks of 32
#define NT 2           // two 16-col N tiles cover 25 (cols 25..31 zero-padded)
#define PACK_ELEMS (KK_ITERS * NT * 32 * 16)   // 25600 bf16 elems per (hi|lo) array

// Pre-pack W1 into the CDNA5 WMMA B-operand lane layout (16-bit B, 32x16):
//   lane L: N = 16*nt + (L&15), K base = (L<16 ? 0 : 8), elem e <-> K = kbase+e (e<8), kbase+16+(e-8) (e>=8)
// Split each f32 into bf16 hi (truncate) + bf16 lo (residual).
__global__ void snn_pack_w1(const float* __restrict__ W1,
                            __bf16* __restrict__ wh, __bf16* __restrict__ wl) {
    int idx = blockIdx.x * blockDim.x + threadIdx.x;
    if (idx >= PACK_ELEMS) return;
    int e    = idx & 15;
    int lane = (idx >> 4) & 31;
    int nt   = (idx >> 9) & 1;
    int kk   = idx >> 10;
    int n = nt * 16 + (lane & 15);
    int k = kk * 32 + ((lane < 16) ? 0 : 8) + ((e < 8) ? e : 8 + e);
    float v  = (n < NH && k < KDIM) ? W1[n * KDIM + k] : 0.0f;
    float hf = __uint_as_float(__float_as_uint(v) & 0xFFFF0000u);
    wh[idx] = (__bf16)hf;
    wl[idx] = (__bf16)(v - hf);
}

__global__ void __launch_bounds__(256)
snn_fused(const float* __restrict__ x,
          const __bf16* __restrict__ wh, const __bf16* __restrict__ wl,
          const float* __restrict__ b1, const float* __restrict__ W2,
          const float* __restrict__ b2,
          float* __restrict__ out_spk, float* __restrict__ out_mem, int Bn) {
    __shared__ float lds_cur[ROWS_PER_BLOCK * LDS_STRIDE];  // cur1 tile [256][25]
    __shared__ float lds_w2[NH * NO];                       // W2 transposed [j][i]

    const int tid  = threadIdx.x;
    const int lane = tid & 31;
    const int wv   = tid >> 5;

    // Stage W2^T into LDS (W2 is [10][25] row-major; we want [25][10]).
    if (tid < NH * NO) {
        int i = tid / NH;          // output neuron
        int j = tid - i * NH;      // hidden neuron
        lds_w2[j * NO + i] = W2[tid];
    }

    // ---------------- Phase 1: WMMA GEMM -----------------------------------
    // Wave wv owns rows [wv*32, wv*32+32) of this block's 256-row batch tile,
    // as two M=16 tiles; N covered by two 16-wide tiles (only 25 cols real).
    const int rowblock = blockIdx.x * ROWS_PER_BLOCK;
    const int mrow  = lane & 15;
    const int khalf = (lane < 16) ? 0 : 8;   // A-operand K sub-half per lane group

    v8f c[2][2];
    #pragma unroll
    for (int m = 0; m < 2; ++m) { c[m][0] = {}; c[m][1] = {}; }

    for (int kk = 0; kk < KK_ITERS; ++kk) {
        const int k0 = kk * 32;
        // B tiles: prepacked, two b128 loads each (hits WGP$/L2).
        const v16bf bh0 = *(const v16bf*)(wh + ((kk * 2 + 0) * 32 + lane) * 16);
        const v16bf bl0 = *(const v16bf*)(wl + ((kk * 2 + 0) * 32 + lane) * 16);
        const v16bf bh1 = *(const v16bf*)(wh + ((kk * 2 + 1) * 32 + lane) * 16);
        const v16bf bl1 = *(const v16bf*)(wl + ((kk * 2 + 1) * 32 + lane) * 16);

        #pragma unroll
        for (int m = 0; m < 2; ++m) {
            // A-operand (16-bit A 16x32): lane holds row mrow,
            // elems 0..7 <-> K = khalf+0..7, elems 8..15 <-> K = khalf+16+0..7.
            const int row = rowblock + wv * 32 + m * 16 + mrow;
            const float* xr = x + (size_t)row * KDIM + k0 + khalf;
            union { float4 q[4]; float f[16]; } au;
            au.q[0] = *(const float4*)(xr + 0);
            au.q[1] = *(const float4*)(xr + 4);
            if (kk < KK_ITERS - 1) {            // uniform guard: last chunk K=784.. pads 0
                au.q[2] = *(const float4*)(xr + 16);
                au.q[3] = *(const float4*)(xr + 20);
            } else {
                au.q[2] = make_float4(0.f, 0.f, 0.f, 0.f);
                au.q[3] = make_float4(0.f, 0.f, 0.f, 0.f);
            }
            v16bf ah, al;
            #pragma unroll
            for (int e = 0; e < 16; ++e) {
                float f  = au.f[e];
                float hf = __uint_as_float(__float_as_uint(f) & 0xFFFF0000u);
                ah[e] = (__bf16)hf;
                al[e] = (__bf16)(f - hf);
            }
            // 3-term hi/lo product: (ah+al)(bh+bl) ~= ah*bh + ah*bl + al*bh
            c[m][0] = __builtin_amdgcn_wmma_f32_16x16x32_bf16(false, ah, false, bh0, (short)0, c[m][0], false, false);
            c[m][0] = __builtin_amdgcn_wmma_f32_16x16x32_bf16(false, ah, false, bl0, (short)0, c[m][0], false, false);
            c[m][0] = __builtin_amdgcn_wmma_f32_16x16x32_bf16(false, al, false, bh0, (short)0, c[m][0], false, false);
            c[m][1] = __builtin_amdgcn_wmma_f32_16x16x32_bf16(false, ah, false, bh1, (short)0, c[m][1], false, false);
            c[m][1] = __builtin_amdgcn_wmma_f32_16x16x32_bf16(false, ah, false, bl1, (short)0, c[m][1], false, false);
            c[m][1] = __builtin_amdgcn_wmma_f32_16x16x32_bf16(false, al, false, bh1, (short)0, c[m][1], false, false);
        }
    }

    // C layout: lane holds col (lane&15), rows r + (lane<16?0:8), r=0..7 per VGPR.
    {
        const int col   = lane & 15;
        const int rbase = (lane < 16) ? 0 : 8;
        const float b1a = b1[col];
        const bool  hi_ok = (col + 16) < NH;
        const float b1b = hi_ok ? b1[col + 16] : 0.0f;
        #pragma unroll
        for (int m = 0; m < 2; ++m) {
            #pragma unroll
            for (int r = 0; r < 8; ++r) {
                int rl = wv * 32 + m * 16 + rbase + r;
                lds_cur[rl * LDS_STRIDE + col] = c[m][0][r] + b1a;
                if (hi_ok) lds_cur[rl * LDS_STRIDE + col + 16] = c[m][1][r] + b1b;
            }
        }
    }
    __syncthreads();

    // ---------------- Phase 2: LIF recurrence, one batch row per thread ----
    const int brow = rowblock + tid;
    float cur[NH], mem1[NH], mem2[NO], b2r[NO];
    #pragma unroll
    for (int j = 0; j < NH; ++j) { cur[j] = lds_cur[tid * LDS_STRIDE + j]; mem1[j] = 0.0f; }
    #pragma unroll
    for (int i = 0; i < NO; ++i) { mem2[i] = 0.0f; b2r[i] = b2[i]; }

    #pragma unroll 1
    for (int t = 0; t < TSTEPS; ++t) {
        float cur2[NO];
        #pragma unroll
        for (int i = 0; i < NO; ++i) cur2[i] = b2r[i];

        #pragma unroll
        for (int j = 0; j < NH; ++j) {
            float r1 = (mem1[j] > 1.0f) ? 1.0f : 0.0f;       // reset from OLD mem
            float m1 = fmaf(BETA, mem1[j], cur[j]) - r1;     // subtract-reset
            mem1[j]  = m1;
            float sj = (m1 > 1.0f) ? 1.0f : 0.0f;            // spike from NEW mem
            #pragma unroll
            for (int i = 0; i < NO; ++i)
                cur2[i] = fmaf(sj, lds_w2[j * NO + i], cur2[i]);
        }

        float spkv[NO];
        #pragma unroll
        for (int i = 0; i < NO; ++i) {
            float r2 = (mem2[i] > 1.0f) ? 1.0f : 0.0f;
            float m2 = fmaf(BETA, mem2[i], cur2[i]) - r2;
            mem2[i]  = m2;
            spkv[i]  = (m2 > 1.0f) ? 1.0f : 0.0f;
        }

        const size_t base = ((size_t)t * Bn + brow) * NO;    // 40B rows, 8B aligned
        float2* os = (float2*)(out_spk + base);
        float2* om = (float2*)(out_mem + base);
        #pragma unroll
        for (int p = 0; p < NO / 2; ++p) {
            os[p] = make_float2(spkv[2 * p], spkv[2 * p + 1]);
            om[p] = make_float2(mem2[2 * p], mem2[2 * p + 1]);
        }
    }
}

extern "C" void kernel_launch(void* const* d_in, const int* in_sizes, int n_in,
                              void* d_out, int out_size, void* d_ws, size_t ws_size,
                              hipStream_t stream) {
    const float* x  = (const float*)d_in[0];
    const float* W1 = (const float*)d_in[1];
    const float* b1 = (const float*)d_in[2];
    const float* W2 = (const float*)d_in[3];
    const float* b2 = (const float*)d_in[4];
    const int Bn = in_sizes[0] / KDIM;          // 32768

    __bf16* wh = (__bf16*)d_ws;                 // 51200 B
    __bf16* wl = wh + PACK_ELEMS;               // +51200 B (100KB total scratch)

    float* out_spk = (float*)d_out;
    float* out_mem = out_spk + (size_t)TSTEPS * Bn * NO;

    snn_pack_w1<<<(PACK_ELEMS + 255) / 256, 256, 0, stream>>>(W1, wh, wl);
    snn_fused<<<Bn / ROWS_PER_BLOCK, 256, 0, stream>>>(x, wh, wl, b1, W2, b2,
                                                       out_spk, out_mem, Bn);
}
